// BF16IndexerBaseline_57612691308721
// MI455X (gfx1250) — matrix operand order repaired and checked by reading, not compiled, not run on hardware
//
#include <hip/hip_runtime.h>
#include <hip/hip_bf16.h>

typedef __bf16 bf16_t;
typedef __attribute__((ext_vector_type(16))) __bf16 bf16x16;
typedef __attribute__((ext_vector_type(8)))  __bf16 bf16x8;
typedef __attribute__((ext_vector_type(8)))  float  f32x8;
typedef int v4i __attribute__((vector_size(4 * sizeof(int))));

#define M_DIM 2048
#define H_DIM 16
#define N_DIM 4096
#define D_DIM 128
#define BM 64
#define BN 64
#define PAD 8
#define LQ_STRIDE (D_DIM + PAD)   // 136 elements -> 272B row, rotates LDS banks
#define SOFTMAX_SCALE 0.08838834764831845f  // 128^-0.5

#if __has_builtin(__builtin_amdgcn_global_load_async_to_lds_b128)
#define ASYNC_LOADS 1
#else
#define ASYNC_LOADS 0
#endif

typedef __attribute__((address_space(1))) v4i* as1_v4i_ptr;
typedef __attribute__((address_space(3))) v4i* as3_v4i_ptr;

static __device__ __forceinline__ as1_v4i_ptr to_as1(const void* p) {
  return (as1_v4i_ptr)(unsigned long long)p;
}
static __device__ __forceinline__ as3_v4i_ptr to_as3(void* p) {
  // generic LDS pointer: low 32 bits are the LDS offset
  return (as3_v4i_ptr)(unsigned int)(unsigned long long)p;
}

// Stage one head's 64x128 bf16 q tile into LDS (16KB -> 4 x b128 per thread).
static __device__ __forceinline__ void load_q_tile(const bf16_t* __restrict__ q,
                                                   int h, int m0, int tid,
                                                   bf16_t* __restrict__ dst) {
#pragma unroll
  for (int i = 0; i < 4; ++i) {
    const int id  = tid + i * 256;
    const int row = id >> 4;      // 0..63
    const int c   = id & 15;      // 16B chunk within the 256B row
    const bf16_t* gp = q + ((size_t)(m0 + row) * H_DIM + h) * D_DIM + c * 8;
    bf16_t* lp = dst + row * LQ_STRIDE + c * 8;
#if ASYNC_LOADS
    __builtin_amdgcn_global_load_async_to_lds_b128(to_as1(gp), to_as3(lp), 0, 0);
#else
    *(uint4*)lp = *(const uint4*)gp;
#endif
  }
}

static __device__ __forceinline__ void qwait_barrier() {
#if ASYNC_LOADS
#if __has_builtin(__builtin_amdgcn_s_wait_asynccnt)
  __builtin_amdgcn_s_wait_asynccnt(0);
#else
  asm volatile("s_wait_asynccnt 0x0" ::: "memory");
#endif
#endif
  __syncthreads();
}

__global__ __launch_bounds__(256)
void bf16_indexer_wmma_kernel(const bf16_t* __restrict__ q,
                              const bf16_t* __restrict__ k,
                              const bf16_t* __restrict__ w,
                              float* __restrict__ out) {
  __shared__ alignas(16) bf16_t lq[2][BM * LQ_STRIDE];   // ~34.8 KB double-buffered q

  const int tid  = threadIdx.x;
  const int wave = tid >> 5;
  const int lane = tid & 31;
  const int half = lane >> 4;   // which 16-lane half
  const int l16  = lane & 15;
  const int mSub = wave >> 1;   // 0..3 : which 16-row band
  const int nGrp = wave & 1;    // 0..1 : which 32-col group
  const int m0 = blockIdx.x * BM;
  const int n0 = blockIdx.y * BN;

  // ---- B fragments (k) straight from global into registers; reused by all 16 heads.
  // B is KxN: lane holds column n = l16 of the tile; half 0 -> K {0..7, 16..23}, half 1 -> K {8..15, 24..31}.
  bf16x16 bfr[2][4];
#pragma unroll
  for (int t = 0; t < 2; ++t) {
    const int n = n0 + nGrp * 32 + t * 16 + l16;
#pragma unroll
    for (int kk = 0; kk < 4; ++kk) {
      const bf16_t* p = k + (size_t)n * D_DIM + kk * 32 + half * 8;
      bf16x8 lo = *(const bf16x8*)p;
      bf16x8 hi = *(const bf16x8*)(p + 16);
      bfr[t][kk] = __builtin_shufflevector(lo, hi,
          0, 1, 2, 3, 4, 5, 6, 7, 8, 9, 10, 11, 12, 13, 14, 15);
    }
  }

  load_q_tile(q, 0, m0, tid, lq[0]);
  qwait_barrier();

  const f32x8 zero8 = {0.f, 0.f, 0.f, 0.f, 0.f, 0.f, 0.f, 0.f};
  f32x8 sum[2];
  sum[0] = zero8;
  sum[1] = zero8;

  // Base of this wave's A-fragment row in LDS (per-buffer offset added in-loop).
  const int aRow = (mSub * 16 + l16) * LQ_STRIDE + half * 8;

  // unroll exactly 2: double-buffer index becomes compile-time static,
  // registers stay well under 256 (no s_set_vgpr_msb churn, better occupancy).
#pragma unroll 2
  for (int h = 0; h < H_DIM; ++h) {
    const int buf = h & 1;
    if (h + 1 < H_DIM) load_q_tile(q, h + 1, m0, tid, lq[buf ^ 1]);  // overlap next head

    // per-row weights for this head: rows m0 + mSub*16 + half*8 + v  (contiguous in m)
    const bf16_t* wp = w + (size_t)h * M_DIM + m0 + mSub * 16 + half * 8;
    bf16x8 wv = *(const bf16x8*)wp;

    f32x8 acc[2];
    acc[0] = zero8;
    acc[1] = zero8;

#pragma unroll
    for (int kk = 0; kk < 4; ++kk) {
      // A fragment from LDS: lane holds row m = mSub*16 + l16;
      // half 0 -> K {0..7, 16..23}, half 1 -> K {8..15, 24..31} (within this 32-chunk)
      const bf16_t* ap = &lq[buf][aRow + kk * 32];
      bf16x8 alo = *(const bf16x8*)ap;
      bf16x8 ahi = *(const bf16x8*)(ap + 16);
      bf16x16 afrag = __builtin_shufflevector(alo, ahi,
          0, 1, 2, 3, 4, 5, 6, 7, 8, 9, 10, 11, 12, 13, 14, 15);
#pragma unroll
      for (int t = 0; t < 2; ++t) {
        acc[t] = __builtin_amdgcn_wmma_f32_16x16x32_bf16(
            false, afrag, false, bfr[t][kk], (short)0, acc[t], false, false);
      }
    }

    // relu + per-row weight, accumulate over heads (scale folded into the store)
#pragma unroll
    for (int t = 0; t < 2; ++t) {
#pragma unroll
      for (int v = 0; v < 8; ++v) {
        sum[t][v] += __builtin_fmaxf(acc[t][v], 0.f) * (float)wv[v];
      }
    }

    if (h + 1 < H_DIM) qwait_barrier();
  }

  // ---- store: C/D layout row m = mSub*16 + half*8 + v, col n = nGrp*32 + t*16 + l16
#pragma unroll
  for (int v = 0; v < 8; ++v) {
    const int m = m0 + mSub * 16 + half * 8 + v;
#pragma unroll
    for (int t = 0; t < 2; ++t) {
      const int n = n0 + nGrp * 32 + t * 16 + l16;
      __builtin_nontemporal_store(sum[t][v] * SOFTMAX_SCALE, out + (size_t)m * N_DIM + n);
    }
  }
}

extern "C" void kernel_launch(void* const* d_in, const int* in_sizes, int n_in,
                              void* d_out, int out_size, void* d_ws, size_t ws_size,
                              hipStream_t stream) {
  const bf16_t* q = (const bf16_t*)d_in[0];   // (1, 2048, 16, 128) bf16
  const bf16_t* k = (const bf16_t*)d_in[1];   // (1, 4096, 128)     bf16
  const bf16_t* w = (const bf16_t*)d_in[2];   // (16, 2048)         bf16
  float* out = (float*)d_out;                 // (1, 2048, 4096)    fp32

  dim3 grid(M_DIM / BM, N_DIM / BN);          // 32 x 64 blocks
  bf16_indexer_wmma_kernel<<<grid, 256, 0, stream>>>(q, k, w, out);
}
